// LinearAttention_57724360458394
// MI455X (gfx1250) — matrix operand order repaired and checked by reading, not compiled, not run on hardware
//
#include <hip/hip_runtime.h>
#include <hip/hip_bf16.h>
#include <stdint.h>

typedef __attribute__((ext_vector_type(16))) __bf16 v16bf;
typedef __attribute__((ext_vector_type(8)))  __bf16 v8bf;
typedef __attribute__((ext_vector_type(8)))  float  v8f;
typedef __attribute__((ext_vector_type(4)))  int    v4i;

#define BATCH 2
#define CHANS 64
#define HEADS 4
#define DHEAD 32
#define CTX_CHUNK 4096

#if defined(__has_builtin)
#if __has_builtin(__builtin_amdgcn_global_load_async_to_lds_b128)
#define HAVE_ASYNC_LDS 1
#endif
#endif

#ifdef HAVE_ASYNC_LDS
__device__ __forceinline__ void async_wait0() {
#if __has_builtin(__builtin_amdgcn_s_wait_asynccnt)
    __builtin_amdgcn_s_wait_asynccnt(0);
#else
    asm volatile("s_wait_asynccnt 0x0" ::: "memory");
#endif
}
#endif

// Load a 16x32 bf16 A-fragment row slice from fp32 memory (this code's contiguous k order, not the ISA table order):
// lane row = lane&15; this lane's K values = ko+{0..15}.
__device__ __forceinline__ v16bf load_a_f32(const float* __restrict__ ap, int ko)
{
    const float4 a0 = *(const float4*)(ap + ko);
    const float4 a1 = *(const float4*)(ap + ko + 4);
    const float4 a2 = *(const float4*)(ap + ko + 8);
    const float4 a3 = *(const float4*)(ap + ko + 12);
    v16bf af;
    af[0]  = (__bf16)a0.x; af[1]  = (__bf16)a0.y; af[2]  = (__bf16)a0.z; af[3]  = (__bf16)a0.w;
    af[4]  = (__bf16)a1.x; af[5]  = (__bf16)a1.y; af[6]  = (__bf16)a1.z; af[7]  = (__bf16)a1.w;
    af[8]  = (__bf16)a2.x; af[9]  = (__bf16)a2.y; af[10] = (__bf16)a2.z; af[11] = (__bf16)a2.w;
    af[12] = (__bf16)a3.x; af[13] = (__bf16)a3.y; af[14] = (__bf16)a3.z; af[15] = (__bf16)a3.w;
    return af;
}

// ---------------------------------------------------------------------------
// Generic bf16-WMMA GEMM: C[M x N] = A[M x K] * B[K x N] (+bias per row)
// Block = 256 threads = 8 waves, 4 (M) x 2 (N) 16x16 tiles.
// Grid: (N/32, M/64, batch).
// ---------------------------------------------------------------------------
template <typename TB, typename TC>
__global__ __launch_bounds__(256) void wmma_gemm(
    const float* __restrict__ A, size_t strideA,
    const TB* __restrict__ B, size_t strideB,
    TC* __restrict__ C, size_t strideC,
    int K, int lda, int ldb, int ldc,
    const float* __restrict__ bias)
{
    const int z = blockIdx.z;
    const float* Ab = A + strideA * (size_t)z;
    const TB*    Bb = B + strideB * (size_t)z;
    TC*          Cb = C + strideC * (size_t)z;

    const int lane = threadIdx.x & 31;
    const int w    = threadIdx.x >> 5;
    const int row0 = blockIdx.y * 64 + (w >> 1) * 16;
    const int col0 = blockIdx.x * 32 + (w & 1) * 16;

    const int arow = row0 + (lane & 15);
    const int ko   = (lane >> 4) * 16;    // A-fragment K base
    const int bcol = col0 + (lane & 15);
    const int kb   = (lane >> 4) * 16;    // B-fragment K base

    v8f acc = {};
    for (int k0 = 0; k0 < K; k0 += 32) {
        const v16bf af = load_a_f32(Ab + (size_t)arow * lda + k0, ko);

        v16bf bf;
        const TB* bp = Bb + (size_t)(k0 + kb) * ldb + bcol;
#pragma unroll
        for (int e = 0; e < 16; ++e) bf[e] = (__bf16)(float)bp[(size_t)e * ldb];

        if (k0 + 32 < K)  // gfx1250 global_prefetch_b8 on next B panel
            __builtin_prefetch(bp + (size_t)32 * ldb, 0, 1);

        acc = __builtin_amdgcn_wmma_f32_16x16x32_bf16(
            false, af, false, bf, (short)0, acc, false, false);
    }

#pragma unroll
    for (int r = 0; r < 8; ++r) {
        const int rr = row0 + (lane >> 4) * 8 + r;
        float v = acc[r];
        if (bias) v += bias[rr];
        Cb[(size_t)rr * ldc + bcol] = (TC)v;
    }
}

// ---------------------------------------------------------------------------
// Row softmax statistics over k = qkv rows [128..255] per batch (bf16 input).
// One block per row; stats[row] = {rowmax, 1/sum(exp(x-rowmax))}.
// ---------------------------------------------------------------------------
__global__ __launch_bounds__(256) void k_row_stats(
    const __bf16* __restrict__ qkv, int N, float* __restrict__ stats)
{
    __shared__ float red[256];
    const int row = blockIdx.x;               // 0 .. BATCH*128-1
    const int b   = row >> 7;
    const int r   = row & 127;
    const __bf16* p = qkv + ((size_t)b * 384 + 128 + r) * (size_t)N;

    const int nvec = N >> 3;                  // 16-byte chunks
    float m = -3.402823466e+38f;
    for (int i = threadIdx.x; i < nvec; i += blockDim.x) {
        const v8bf c = *(const v8bf*)(p + (size_t)i * 8);
#pragma unroll
        for (int e = 0; e < 8; ++e) m = fmaxf(m, (float)c[e]);
    }
    for (int i = (nvec << 3) + threadIdx.x; i < N; i += blockDim.x)
        m = fmaxf(m, (float)p[i]);
    red[threadIdx.x] = m;
    __syncthreads();
    for (int s = 128; s > 0; s >>= 1) {
        if ((int)threadIdx.x < s)
            red[threadIdx.x] = fmaxf(red[threadIdx.x], red[threadIdx.x + s]);
        __syncthreads();
    }
    m = red[0];
    __syncthreads();

    float sum = 0.f;
    for (int i = threadIdx.x; i < nvec; i += blockDim.x) {
        const v8bf c = *(const v8bf*)(p + (size_t)i * 8);
#pragma unroll
        for (int e = 0; e < 8; ++e) sum += __expf((float)c[e] - m);
    }
    for (int i = (nvec << 3) + threadIdx.x; i < N; i += blockDim.x)
        sum += __expf((float)p[i] - m);
    red[threadIdx.x] = sum;
    __syncthreads();
    for (int s = 128; s > 0; s >>= 1) {
        if ((int)threadIdx.x < s)
            red[threadIdx.x] += red[threadIdx.x + s];
        __syncthreads();
    }
    if (threadIdx.x == 0) {
        stats[row * 2]     = m;
        stats[row * 2 + 1] = 1.0f / red[0];   // reciprocal: consumers multiply
    }
}

// ---------------------------------------------------------------------------
// context[b,h,d,e] = sum_n softmax(k)[d,n] * v[e,n], via bf16 WMMA.
// exp(k - rowmax) built into the A fragment; 1/Z applied at atomic store.
// Block = 128 threads = 4 waves (2x2 tiles).  Grid: (N/CTX_CHUNK, HEADS, B).
// ---------------------------------------------------------------------------
__global__ __launch_bounds__(128) void k_context(
    const __bf16* __restrict__ qkv, const float* __restrict__ stats,
    float* __restrict__ ctx, int N)
{
    const int b = blockIdx.z, h = blockIdx.y;
    const int n0base = blockIdx.x * CTX_CHUNK;
    const int lane = threadIdx.x & 31;
    const int w    = threadIdx.x >> 5;      // 0..3
    const int dbase = (w >> 1) * 16;
    const int ebase = (w & 1) * 16;

    const __bf16* kmat = qkv + ((size_t)b * 384 + 128 + h * DHEAD) * (size_t)N;
    const __bf16* vmat = qkv + ((size_t)b * 384 + 256 + h * DHEAD) * (size_t)N;

    const int drow = dbase + (lane & 15);
    const float mrow = stats[(b * 128 + h * DHEAD + drow) * 2];
    const int ko = (lane >> 4) * 16;
    const int kb = (lane >> 4) * 16;
    const __bf16* krow = kmat + (size_t)drow * N;

    const int ecol = ebase + (lane & 15);
    const __bf16* vrow = vmat + (size_t)ecol * N;

    v8f acc = {};
    const int n1 = (n0base + CTX_CHUNK < N) ? (n0base + CTX_CHUNK) : N;
    for (int n0 = n0base; n0 < n1; n0 += 32) {
        const v8bf k0v = *(const v8bf*)(krow + n0 + ko);
        const v8bf k1v = *(const v8bf*)(krow + n0 + ko + 8);
        const v8bf b0v = *(const v8bf*)(vrow + n0 + kb);
        const v8bf b1v = *(const v8bf*)(vrow + n0 + kb + 8);

        v16bf af, bf;
#pragma unroll
        for (int e = 0; e < 8; ++e) {
            af[e]     = (__bf16)__expf((float)k0v[e] - mrow);
            af[8 + e] = (__bf16)__expf((float)k1v[e] - mrow);
            bf[e]     = b0v[e];
            bf[8 + e] = b1v[e];
        }

        acc = __builtin_amdgcn_wmma_f32_16x16x32_bf16(
            false, af, false, bf, (short)0, acc, false, false);
    }

    const int col = lane & 15;
#pragma unroll
    for (int r = 0; r < 8; ++r) {
        const int d = dbase + (lane >> 4) * 8 + r;
        const float invZ = stats[(b * 128 + h * DHEAD + d) * 2 + 1];
        atomicAdd(&ctx[(((size_t)b * HEADS + h) * DHEAD + d) * DHEAD + ebase + col],
                  acc[r] * invZ);
    }
}

// ---------------------------------------------------------------------------
// Fold w_out into context:  M[b][o][h*32+d] = sum_e w_out[o][h*32+e]*ctx[b,h,d,e]
// ---------------------------------------------------------------------------
__global__ __launch_bounds__(256) void fold_ctx(
    const float* __restrict__ ctx, const float* __restrict__ w_out,
    float* __restrict__ Mmat)
{
    const int b = blockIdx.x;
    for (int idx = threadIdx.x; idx < 64 * 128; idx += blockDim.x) {
        const int o  = idx >> 7;
        const int hd = idx & 127;
        const int h  = hd >> 5;
        const int d  = hd & 31;
        float s = 0.f;
        const float* wrow = w_out + o * 128 + h * DHEAD;
        const float* crow = ctx + (((size_t)b * HEADS + h) * DHEAD + d) * DHEAD;
#pragma unroll
        for (int e = 0; e < DHEAD; ++e) s += wrow[e] * crow[e];
        Mmat[(size_t)b * 64 * 128 + idx] = s;
    }
}

#ifdef HAVE_ASYNC_LDS
// ---------------------------------------------------------------------------
// Specialized output GEMM: y[64 x N] = Mfold[64x128] @ q[128 x N] + b_out.
// Stages the 128x32 bf16 B panel into LDS with global_load_async_to_lds_b128
// (ASYNCcnt), then feeds WMMA fragments from LDS.  K fully unrolled (4 WMMA).
// Grid: (N/32, 1, batch), 256 threads.
// ---------------------------------------------------------------------------
__global__ __launch_bounds__(256) void wmma_gemm_out_async(
    const float* __restrict__ A, size_t strideA,
    const __bf16* __restrict__ B, size_t strideB,
    float* __restrict__ C, size_t strideC,
    int ldb, int ldc, const float* __restrict__ bias)
{
    __shared__ __bf16 tile[128 * 32];   // 8 KB B panel, row-major [k][col]

    const int z = blockIdx.z;
    const float*  Ab = A + strideA * (size_t)z;
    const __bf16* Bb = B + strideB * (size_t)z + (size_t)blockIdx.x * 32;
    float*        Cb = C + strideC * (size_t)z;

    // Async-stage B panel: 512 x 16B chunks, 2 per thread.
    const int t = threadIdx.x;
#pragma unroll
    for (int i = 0; i < 2; ++i) {
        const int c    = t + i * 256;
        const int krow = c >> 2;
        const int coff = (c & 3) * 8;
        const __bf16* g = Bb + (size_t)krow * ldb + coff;
        __bf16* l = &tile[krow * 32 + coff];
        __builtin_amdgcn_global_load_async_to_lds_b128((v4i*)g, (v4i*)l, 0, 0);
    }
    async_wait0();
    __syncthreads();

    const int lane = threadIdx.x & 31;
    const int w    = threadIdx.x >> 5;
    const int row0 = (w >> 1) * 16;                    // M tile (M = 64)
    const int colT = (w & 1) * 16 + (lane & 15);       // column within panel
    const int arow = row0 + (lane & 15);
    const int ko   = (lane >> 4) * 16;
    const int kb   = (lane >> 4) * 16;

    v8f acc = {};
#pragma unroll
    for (int k0 = 0; k0 < 128; k0 += 32) {
        const v16bf af = load_a_f32(Ab + (size_t)arow * 128 + k0, ko);
        v16bf bf;
#pragma unroll
        for (int e = 0; e < 16; ++e) bf[e] = tile[(k0 + kb + e) * 32 + colT];
        acc = __builtin_amdgcn_wmma_f32_16x16x32_bf16(
            false, af, false, bf, (short)0, acc, false, false);
    }

    const int gcol = blockIdx.x * 32 + colT;
#pragma unroll
    for (int r = 0; r < 8; ++r) {
        const int rr = row0 + (lane >> 4) * 8 + r;
        Cb[(size_t)rr * ldc + gcol] = acc[r] + bias[rr];
    }
}
#endif

// ---------------------------------------------------------------------------
extern "C" void kernel_launch(void* const* d_in, const int* in_sizes, int n_in,
                              void* d_out, int out_size, void* d_ws, size_t ws_size,
                              hipStream_t stream)
{
    (void)n_in; (void)out_size; (void)ws_size;
    const float* x     = (const float*)d_in[0];   // (2, 64, N)
    const float* w_qkv = (const float*)d_in[1];   // (384, 64)
    const float* w_out = (const float*)d_in[2];   // (64, 128)
    const float* b_out = (const float*)d_in[3];   // (64,)
    float* y = (float*)d_out;                     // (2, 64, N)

    const int N = in_sizes[0] / (BATCH * CHANS);  // 110592

    // Workspace layout
    __bf16* qkv    = (__bf16*)d_ws;                         // 2*384*N bf16
    float*  tail   = (float*)(qkv + (size_t)BATCH * 384 * N);
    float*  stats  = tail;                                  // 256*2
    float*  ctx    = stats + BATCH * 128 * 2;               // 2*4*32*32
    float*  Mmat   = ctx + BATCH * HEADS * DHEAD * DHEAD;   // 2*64*128

    // 1) qkv = w_qkv @ x   (fp32 in, bf16 out)
    wmma_gemm<float, __bf16><<<dim3(N / 32, 384 / 64, BATCH), 256, 0, stream>>>(
        w_qkv, 0, x, (size_t)CHANS * N, qkv, (size_t)384 * N,
        /*K=*/64, /*lda=*/64, /*ldb=*/N, /*ldc=*/N, nullptr);

    // 2) softmax row statistics for k (max and 1/Z)
    k_row_stats<<<BATCH * 128, 256, 0, stream>>>(qkv, N, stats);

    // 3) zero ctx accumulators (fresh every launch)
    (void)hipMemsetAsync(ctx, 0, sizeof(float) * BATCH * HEADS * DHEAD * DHEAD, stream);

    // 4) context = softmax(k) @ v^T
    k_context<<<dim3((N + CTX_CHUNK - 1) / CTX_CHUNK, HEADS, BATCH), 128, 0, stream>>>(
        qkv, stats, ctx, N);

    // 5) fold w_out into context -> M[b] (64x128)
    fold_ctx<<<BATCH, 256, 0, stream>>>(ctx, w_out, Mmat);

    // 6) y = M @ q + b_out   (q = qkv rows 0..127 of each batch)
#ifdef HAVE_ASYNC_LDS
    wmma_gemm_out_async<<<dim3(N / 32, 1, BATCH), 256, 0, stream>>>(
        Mmat, (size_t)64 * 128, qkv, (size_t)384 * N, y, (size_t)CHANS * N,
        /*ldb=*/N, /*ldc=*/N, b_out);
#else
    wmma_gemm<__bf16, float><<<dim3(N / 32, 1, BATCH), 256, 0, stream>>>(
        Mmat, (size_t)64 * 128, qkv, (size_t)384 * N, y, (size_t)CHANS * N,
        /*K=*/128, /*lda=*/128, /*ldb=*/N, /*ldc=*/N, b_out);
#endif
}
